// MemoryNetwork_6528350290117
// MI455X (gfx1250) — matrix operand order repaired and checked by reading, not compiled, not run on hardware
//
#include <hip/hip_runtime.h>
#include <hip/hip_bf16.h>
#include <math.h>

typedef __attribute__((ext_vector_type(16))) __bf16 v16bf;
typedef __attribute__((ext_vector_type(8)))  __bf16 v8bf;
typedef __attribute__((ext_vector_type(8)))  float  v8f;
typedef int v4i __attribute__((vector_size(16)));

#define B_ROWS   32768
#define KDIM     768
#define NPAD     192      // 180 useful cols (S:0..89, G:96..185) padded to 12x16
#define TILE_M   128
#define KCHUNK   32
#define NCHUNKS  (KDIM / KCHUNK)   // 24
#define LDA      40       // padded bf16 row stride (bank-conflict-free b128 gathers)
#define LDC      40
#define TAU      32.0f

// Async global->LDS copy path (CDNA5). Fall back to load/store if absent.
#if defined(__has_builtin)
#if __has_builtin(__builtin_amdgcn_global_load_async_to_lds_b128) && \
    __has_builtin(__builtin_amdgcn_s_wait_asynccnt)
#define HAVE_ASYNC 1
#endif
#endif
#ifndef HAVE_ASYNC
#define HAVE_ASYNC 0
#endif

#if HAVE_ASYNC
typedef __attribute__((address_space(1))) v4i gas_v4i;  // global
typedef __attribute__((address_space(3))) v4i las_v4i;  // LDS
#endif

static __device__ __forceinline__ void copy16_g2l(__bf16* l, const __bf16* g) {
#if HAVE_ASYNC
  __builtin_amdgcn_global_load_async_to_lds_b128(
      (gas_v4i*)(void*)const_cast<__bf16*>(g),
      (las_v4i*)(void*)l, 0, 0);
#else
  *(v4i*)(void*)l = *(const v4i*)(const void*)g;
#endif
}

// ---------------------------------------------------------------------------
// Kernel 0: fold weights into memory banks and pre-split to bf16 hi/lo in the
// exact chunk-major padded LDS image:  Cb[(k0/32)*192 + j][40] (cols 0..31 valid)
//   row j in [0,90)   : sum_e DM[j,e]    * W_topic[e,i]
//   row j in [96,186) : sum_e DM[j-96,e] * W_domain[e,i]
//   zeros elsewhere (padding rows)
// ---------------------------------------------------------------------------
__global__ __launch_bounds__(256) void build_comb(const float* __restrict__ Wt,
                                                  const float* __restrict__ Wd,
                                                  const float* __restrict__ DM,
                                                  __bf16* __restrict__ ChiG,
                                                  __bf16* __restrict__ CloG) {
  __shared__ float dmrow[KDIM];
  const int j = blockIdx.x;        // 0..191 (uniform per block)
  const int t = threadIdx.x;
  const bool isS = (j < 90);
  const bool isG = (j >= 96 && j < 186);
  if (!isS && !isG) {
    for (int i = t; i < KDIM; i += 256) {
      const size_t o = (size_t)((i >> 5) * NPAD + j) * LDC + (i & 31);
      ChiG[o] = (__bf16)0.0f;
      CloG[o] = (__bf16)0.0f;
    }
    return;
  }
  const float* dm = isS ? (DM + (size_t)j * KDIM) : (DM + (size_t)(j - 96) * KDIM);
  const float* W  = isS ? Wt : Wd;
  for (int i = t; i < KDIM; i += 256) dmrow[i] = dm[i];
  __syncthreads();
  float s[3] = {0.f, 0.f, 0.f};
  for (int e = 0; e < KDIM; ++e) {
    const float d = dmrow[e];
    const float* wr = W + (size_t)e * KDIM;
    s[0] = fmaf(d, wr[t], s[0]);
    s[1] = fmaf(d, wr[t + 256], s[1]);
    s[2] = fmaf(d, wr[t + 512], s[2]);
  }
  #pragma unroll
  for (int p = 0; p < 3; ++p) {
    const int i = t + p * 256;
    const size_t o = (size_t)((i >> 5) * NPAD + j) * LDC + (i & 31);
    const __bf16 hb = (__bf16)s[p];
    ChiG[o] = hb;
    CloG[o] = (__bf16)(s[p] - (float)hb);
  }
}

// ---------------------------------------------------------------------------
// Kernel 1: per-row 1/||x||.  One wave32 per row, 8 rows per block.
// ---------------------------------------------------------------------------
__global__ __launch_bounds__(256) void row_norms(const float* __restrict__ feat,
                                                 float* __restrict__ inv_norm) {
  const int t    = threadIdx.x;
  const int lane = t & 31;
  const int row  = blockIdx.x * 8 + (t >> 5);
  const float* fr = feat + (size_t)row * KDIM;
  float s = 0.f;
  for (int j = lane; j < KDIM; j += 32) {
    const float v = fr[j];
    s = fmaf(v, v, s);
  }
  #pragma unroll
  for (int off = 16; off > 0; off >>= 1) s += __shfl_xor(s, off, 32);
  if (lane == 0) inv_norm[row] = 1.0f / sqrtf(s);
}

// ---------------------------------------------------------------------------
// helper: assemble v16bf fragment from two LDS b128 reads
// ---------------------------------------------------------------------------
static __device__ __forceinline__ v16bf frag_2x8(const __bf16* p0, const __bf16* p1) {
  v8bf a = *(const v8bf*)p0;
  v8bf b = *(const v8bf*)p1;
  v16bf r;
  #pragma unroll
  for (int i = 0; i < 8; ++i) { r[i] = a[i]; r[i + 8] = b[i]; }
  return r;
}

// ---------------------------------------------------------------------------
// Kernel 2: fused  normalize -> [128x768]x[768x192] split-bf16 WMMA GEMM ->
//           softmax_m -> domain dot -> softmax_d -> out[b,9]
// 256 threads (8 waves). Wave w: M-tiles {w>>1, (w>>1)+4}, N-tiles (w&1)*6..+5.
// 36 WMMAs per wave per K-chunk.
// ---------------------------------------------------------------------------
__global__ __launch_bounds__(256) void fused_gemm(const float* __restrict__ feat,
                                                  const float* __restrict__ inv_norm,
                                                  const __bf16* __restrict__ ChiG,
                                                  const __bf16* __restrict__ CloG,
                                                  float* __restrict__ out) {
  extern __shared__ __align__(16) unsigned char smem[];
  __bf16* Ahi = (__bf16*)smem;                 // 128*LDA
  __bf16* Alo = Ahi + TILE_M * LDA;            // 128*LDA
  __bf16* Chi = Alo + TILE_M * LDA;            // 192*LDC
  __bf16* Clo = Chi + NPAD * LDC;              // 192*LDC
  float*  Sld = (float*)smem;                  // alias: 128*192 fp32 epilogue scratch

  const int t    = threadIdx.x;
  const int w    = t >> 5;
  const int mt0  = w >> 1;          // first M-tile (0..3); second is mt0+4
  const int half = w & 1;           // N-tile group
  const int lane = t & 31;
  const int ln   = lane & 15;
  const int h    = lane >> 4;

  const int lrow = t >> 1;          // 0..127 (cooperative A-loader row)
  const int lc16 = (t & 1) * 16;    // 0 or 16

  const int m0 = blockIdx.x * TILE_M;
  const float invn = inv_norm[m0 + lrow];
  const float* Arow = feat + (size_t)(m0 + lrow) * KDIM;

  v8f acc[2][6];
  #pragma unroll
  for (int g = 0; g < 2; ++g)
    #pragma unroll
    for (int i = 0; i < 6; ++i) acc[g][i] = (v8f){};

  for (int k0 = 0; k0 < KDIM; k0 += KCHUNK) {
    // ---- stage A chunk: 16 floats/thread, normalize + bf16 hi/lo split ----
    __builtin_prefetch(Arow + k0 + KCHUNK + lc16, 0, 1);
    {
      const float* src = Arow + k0 + lc16;
      float v[16];
      #pragma unroll
      for (int q = 0; q < 4; ++q) {
        float4 f = *(const float4*)(src + 4 * q);
        v[4 * q + 0] = f.x; v[4 * q + 1] = f.y; v[4 * q + 2] = f.z; v[4 * q + 3] = f.w;
      }
      v8bf h8[2], l8[2];
      #pragma unroll
      for (int u = 0; u < 16; ++u) {
        const float x  = v[u] * invn;
        const __bf16 hb = (__bf16)x;
        h8[u >> 3][u & 7] = hb;
        l8[u >> 3][u & 7] = (__bf16)(x - (float)hb);
      }
      *(v8bf*)(Ahi + lrow * LDA + lc16)     = h8[0];
      *(v8bf*)(Ahi + lrow * LDA + lc16 + 8) = h8[1];
      *(v8bf*)(Alo + lrow * LDA + lc16)     = l8[0];
      *(v8bf*)(Alo + lrow * LDA + lc16 + 8) = l8[1];
    }
    // ---- stage C chunk: pure 16B async copies of pre-split bf16 panel ----
    {
      const size_t cbase = (size_t)(k0 >> 5) * (NPAD * LDC);
      #pragma unroll
      for (int p = 0; p < 3; ++p) {
        const int idx  = t + p * 256;                    // 0..767
        const int roff = (idx >> 2) * LDC + (idx & 3) * 8;
        copy16_g2l(Chi + roff, ChiG + cbase + roff);
        copy16_g2l(Clo + roff, CloG + cbase + roff);
      }
    }
#if HAVE_ASYNC
    __builtin_amdgcn_s_wait_asynccnt(0);
#endif
    __syncthreads();

    // ---- fragments + 3-term split-bf16 WMMA ----
    // A 16x32 bf16 layout: lane(h,m) holds K = 16h..16h+15 contiguous
    const __bf16* apH0 = Ahi + (mt0 * 16 + ln) * LDA;
    const __bf16* apL0 = Alo + (mt0 * 16 + ln) * LDA;
    const __bf16* apH1 = apH0 + 64 * LDA;                // mt0+4
    const __bf16* apL1 = apL0 + 64 * LDA;
    const v16bf aHi0 = frag_2x8(apH0 + 16 * h, apH0 + 16 * h + 8);
    const v16bf aLo0 = frag_2x8(apL0 + 16 * h, apL0 + 16 * h + 8);
    const v16bf aHi1 = frag_2x8(apH1 + 16 * h, apH1 + 16 * h + 8);
    const v16bf aLo1 = frag_2x8(apL1 + 16 * h, apL1 + 16 * h + 8);

    #pragma unroll
    for (int tt = 0; tt < 6; ++tt) {
      const int nt = half * 6 + tt;
      // B 32x16 bf16 layout: lane(h,n) holds K = 16h..16h+15 contiguous
      const __bf16* bpH = Chi + (nt * 16 + ln) * LDC + 16 * h;
      const __bf16* bpL = Clo + (nt * 16 + ln) * LDC + 16 * h;
      const v16bf bHi = frag_2x8(bpH, bpH + 8);
      const v16bf bLo = frag_2x8(bpL, bpL + 8);
      acc[0][tt] = __builtin_amdgcn_wmma_f32_16x16x32_bf16(
          false, aHi0, false, bHi, (short)0, acc[0][tt], false, false);
      acc[0][tt] = __builtin_amdgcn_wmma_f32_16x16x32_bf16(
          false, aHi0, false, bLo, (short)0, acc[0][tt], false, false);
      acc[0][tt] = __builtin_amdgcn_wmma_f32_16x16x32_bf16(
          false, aLo0, false, bHi, (short)0, acc[0][tt], false, false);
      acc[1][tt] = __builtin_amdgcn_wmma_f32_16x16x32_bf16(
          false, aHi1, false, bHi, (short)0, acc[1][tt], false, false);
      acc[1][tt] = __builtin_amdgcn_wmma_f32_16x16x32_bf16(
          false, aHi1, false, bLo, (short)0, acc[1][tt], false, false);
      acc[1][tt] = __builtin_amdgcn_wmma_f32_16x16x32_bf16(
          false, aLo1, false, bHi, (short)0, acc[1][tt], false, false);
    }
    __syncthreads();
  }

  // ---- spill 128x192 tile into LDS (C/D layout: VGPR r -> M=r+8h, N=ln) ----
  #pragma unroll
  for (int g = 0; g < 2; ++g) {
    const int mrow = (mt0 + 4 * g) * 16 + 8 * h;
    #pragma unroll
    for (int tt = 0; tt < 6; ++tt) {
      const int ncol = (half * 6 + tt) * 16 + ln;
      #pragma unroll
      for (int r = 0; r < 8; ++r) {
        Sld[(mrow + r) * NPAD + ncol] = acc[g][tt][r];
      }
    }
  }
  __syncthreads();

  // ---- epilogue: softmax over m, domain dot, softmax over d ----
  if (t < TILE_M) {
    const float* Sr = Sld + t * NPAD;   // [0..89]=S, [96..185]=G
    float dl[9];
    #pragma unroll
    for (int d = 0; d < 9; ++d) {
      const int base = d * 10;
      float mx = -INFINITY;
      #pragma unroll
      for (int m = 0; m < 10; ++m) mx = fmaxf(mx, Sr[base + m]);
      float se = 0.f, sg = 0.f;
      #pragma unroll
      for (int m = 0; m < 10; ++m) {
        const float e = __expf(TAU * (Sr[base + m] - mx));
        se += e;
        sg = fmaf(e, Sr[96 + base + m], sg);
      }
      dl[d] = TAU * sg / se;
    }
    float mx2 = -INFINITY;
    #pragma unroll
    for (int d = 0; d < 9; ++d) mx2 = fmaxf(mx2, dl[d]);
    float ss = 0.f, ex[9];
    #pragma unroll
    for (int d = 0; d < 9; ++d) { ex[d] = __expf(dl[d] - mx2); ss += ex[d]; }
    const float inv = 1.0f / ss;
    float* op = out + (size_t)(m0 + t) * 9;
    #pragma unroll
    for (int d = 0; d < 9; ++d) op[d] = ex[d] * inv;
  }
}

// ---------------------------------------------------------------------------
extern "C" void kernel_launch(void* const* d_in, const int* in_sizes, int n_in,
                              void* d_out, int out_size, void* d_ws, size_t ws_size,
                              hipStream_t stream) {
  (void)in_sizes; (void)n_in; (void)out_size; (void)ws_size;
  const float* feature = (const float*)d_in[0];
  // d_in[1] = category: unused by the reference computation
  const float* W_topic  = (const float*)d_in[2];
  const float* W_domain = (const float*)d_in[3];
  const float* dom_mem  = (const float*)d_in[4];
  float* out = (float*)d_out;

  float*  inv_norm = (float*)d_ws;                              // 32768 floats
  __bf16* ChiG = (__bf16*)((char*)d_ws + (size_t)B_ROWS * 4);   // 24*192*40 bf16
  __bf16* CloG = ChiG + (size_t)NCHUNKS * NPAD * LDC;

  build_comb<<<NPAD, 256, 0, stream>>>(W_topic, W_domain, dom_mem, ChiG, CloG);
  row_norms<<<B_ROWS / 8, 256, 0, stream>>>(feature, inv_norm);

  const size_t smem_bytes = (size_t)TILE_M * NPAD * sizeof(float);  // 98304
  fused_gemm<<<B_ROWS / TILE_M, 256, smem_bytes, stream>>>(feature, inv_norm,
                                                           ChiG, CloG, out);
}
